// DDiTBlock_26714696581570
// MI455X (gfx1250) — matrix operand order repaired and checked by reading, not compiled, not run on hardware
//
#include <hip/hip_runtime.h>

// ---------------------------------------------------------------------------
// MI455X (gfx1250) fused attention block, bf16 WMMA everywhere.
// Round 4: fix tensor_load_to_lds arity (this toolchain = clang-23 form,
// 6 args: g0 u32x4, g1 i32x8, i32x4, i32x4, i32x8, i32 cpol).
// GEMM tile staging via Tensor Data Mover (tensor_load_to_lds +
// s_wait_tensorcnt, double-buffered LDS); fallback to register pipeline.
// Workspace (bf16): xb[4096,1024] wqb[3072,1024] wob[1024,1024]
//                   qkvb[4096,3072] yb[4096,1024]   (~48 MB total)
// ---------------------------------------------------------------------------

typedef __bf16 bf16;
typedef __attribute__((ext_vector_type(16))) __bf16 v16bf;
typedef __attribute__((ext_vector_type(8)))  float  v8f;
typedef __attribute__((ext_vector_type(4)))  unsigned int u32x4;
typedef __attribute__((ext_vector_type(8)))  int i32x8;
typedef __attribute__((ext_vector_type(4)))  int i32x4;

#if defined(__has_builtin)
#if __has_builtin(__builtin_amdgcn_tensor_load_to_lds) && \
    __has_builtin(__builtin_amdgcn_s_wait_tensorcnt)
#define USE_TDM 1
#endif
#endif

static __device__ __forceinline__ v16bf frag16(const bf16* p0, const bf16* p1) {
  union { v16bf v; uint4 u[2]; } f;
  f.u[0] = *(const uint4*)p0;
  f.u[1] = *(const uint4*)p1;
  return f.v;
}

#ifdef USE_TDM
// TDM 2-D tile load: tile_w contiguous bf16 elements per row, tile_h rows,
// row stride `stride` elements; linear fill into LDS at lds byte offset.
// D# bit layout per CDNA5 ISA ch.8 (group0 128b, group1 256b; 2-D so
// remaining groups are zero).
static __device__ __forceinline__ void tdm_load_2d(
    unsigned lds_byte_off, const bf16* gptr, int tile_w, int tile_h,
    int stride) {
  unsigned long long ga = (unsigned long long)gptr;
  u32x4 g0;
  g0[0] = 1u;                                        // count=1, user mode
  g0[1] = lds_byte_off;                              // lds_addr
  g0[2] = (unsigned)(ga & 0xffffffffu);              // global_addr[31:0]
  g0[3] = (unsigned)((ga >> 32) & 0x1ffffffu)        // global_addr[56:32]
          | (2u << 30);                              // type=2 ("image")
  i32x8 g1;
  g1[0] = 1 << 16;                                   // data_size=1 (2B), mask=0
  g1[1] = (tile_w & 0xffff) << 16;                   // tensor_dim0[15:0]
  g1[2] = ((unsigned)tile_w >> 16)                   // tensor_dim0[31:16]
          | ((tile_h & 0xffff) << 16);               // tensor_dim1[15:0]
  g1[3] = ((unsigned)tile_h >> 16)                   // tensor_dim1[31:16]
          | ((tile_w & 0xffff) << 16);               // tile_dim0
  g1[4] = (tile_h & 0xffff);                         // tile_dim1 (tile_dim2=0)
  g1[5] = stride;                                    // tensor_dim0_stride[31:0]
  g1[6] = 0;                                         // stride[47:32], s1 lo
  g1[7] = 0;                                         // s1 hi
  i32x4 z4 = {0, 0, 0, 0};
  i32x8 z8 = {0, 0, 0, 0, 0, 0, 0, 0};
  __builtin_amdgcn_tensor_load_to_lds(g0, g1, z4, z4, z8, 0);
}

static __device__ __forceinline__ unsigned lds_off(const void* p) {
  // generic LDS address: low 32 bits are the wave-relative LDS byte offset
  return (unsigned)(unsigned long long)p;
}
#endif

// ---------------------------------------------------------------------------
// f32 -> bf16 conversion, 4 elements / thread
// ---------------------------------------------------------------------------
__global__ __launch_bounds__(256) void f32_to_bf16_kernel(
    const float* __restrict__ in, bf16* __restrict__ out, int n) {
  int i = (blockIdx.x * 256 + threadIdx.x) * 4;
  if (i >= n) return;
  float4 f = *(const float4*)(in + i);
  union { bf16 h[4]; unsigned long long u; } r;
  r.h[0] = (bf16)f.x; r.h[1] = (bf16)f.y; r.h[2] = (bf16)f.z; r.h[3] = (bf16)f.w;
  *(unsigned long long*)(out + i) = r.u;
}

// ---------------------------------------------------------------------------
// O[M,N] = A[M,K] * W[N,K]^T, bf16 in, OutT out.
// Block tile 128x128, 8 waves, wave tile 32x64 (2x4 subtiles), K step 32.
// TDM path: double-buffered LDS, wave 0 DMAs stage i+1 while all compute i.
// ---------------------------------------------------------------------------
template <typename OutT>
__global__ __launch_bounds__(256) void gemm_bf16_kernel(
    const bf16* __restrict__ A, const bf16* __restrict__ W,
    OutT* __restrict__ O, int M, int N, int K) {
  const int tid  = threadIdx.x;
  const int wave = tid >> 5, lane = tid & 31;
  const int nlo  = lane & 15, hi = lane >> 4;
  const int m0 = (wave >> 1) * 32;   // 0,32,64,96
  const int n0 = (wave & 1) * 64;    // 0,64
  const long blockM = (long)blockIdx.y * 128;
  const long blockN = (long)blockIdx.x * 128;

  v8f acc[2][4] = {};

#ifdef USE_TDM
  __shared__ bf16 As[2][128 * 32];
  __shared__ bf16 Ws[2][128 * 32];

  if (tid < 32) {  // wave 0 drives the Tensor Data Mover
    tdm_load_2d(lds_off(&As[0][0]), &A[blockM * (long)K], 32, 128, K);
    tdm_load_2d(lds_off(&Ws[0][0]), &W[blockN * (long)K], 32, 128, K);
  }

  int p = 0;
  for (int k0 = 0; k0 < K; k0 += 32, p ^= 1) {
    __syncthreads();  // buffer p^1 no longer being read: safe to overwrite
    if (tid < 32) {
      if (k0 + 32 < K) {
        tdm_load_2d(lds_off(&As[p ^ 1][0]), &A[blockM * (long)K + k0 + 32],
                    32, 128, K);
        tdm_load_2d(lds_off(&Ws[p ^ 1][0]), &W[blockN * (long)K + k0 + 32],
                    32, 128, K);
        __builtin_amdgcn_s_wait_tensorcnt(2);  // in-order: stage p landed
      } else {
        __builtin_amdgcn_s_wait_tensorcnt(0);
      }
    }
    __syncthreads();  // stage p visible to all waves
    const bf16* Ab = &As[p][0];
    const bf16* Wb = &Ws[p][0];
#else
  __shared__ bf16 AsBuf[128 * 32];
  __shared__ bf16 WsBuf[128 * 32];

  const int rowL = tid >> 2;           // 0..63
  const int offL = (tid & 3) * 8;      // 0,8,16,24
  const bf16* gA0 = &A[(blockM + rowL) * (long)K + offL];
  const bf16* gA1 = &A[(blockM + rowL + 64) * (long)K + offL];
  const bf16* gW0 = &W[(blockN + rowL) * (long)K + offL];
  const bf16* gW1 = &W[(blockN + rowL + 64) * (long)K + offL];
  uint4 ra0 = *(const uint4*)gA0;
  uint4 ra1 = *(const uint4*)gA1;
  uint4 rw0 = *(const uint4*)gW0;
  uint4 rw1 = *(const uint4*)gW1;

  for (int k0 = 0; k0 < K; k0 += 32) {
    __syncthreads();
    *(uint4*)&AsBuf[rowL * 32 + offL]        = ra0;
    *(uint4*)&AsBuf[(rowL + 64) * 32 + offL] = ra1;
    *(uint4*)&WsBuf[rowL * 32 + offL]        = rw0;
    *(uint4*)&WsBuf[(rowL + 64) * 32 + offL] = rw1;
    __syncthreads();
    if (k0 + 32 < K) {
      ra0 = *(const uint4*)(gA0 + k0 + 32);
      ra1 = *(const uint4*)(gA1 + k0 + 32);
      rw0 = *(const uint4*)(gW0 + k0 + 32);
      rw1 = *(const uint4*)(gW1 + k0 + 32);
    }
    if (k0 + 64 < K) {
      __builtin_prefetch(gA0 + k0 + 64, 0, 0);
      __builtin_prefetch(gW0 + k0 + 64, 0, 0);
    }
    const bf16* Ab = &AsBuf[0];
    const bf16* Wb = &WsBuf[0];
#endif

    v16bf afrag[2], bfrag[4];
#pragma unroll
    for (int ms = 0; ms < 2; ++ms) {
      // A 16x32 layout: lane = M row, lane[4] picks K 0-7/16-23 vs 8-15/24-31
      const bf16* base = &Ab[(m0 + ms * 16 + nlo) * 32 + hi * 8];
      afrag[ms] = frag16(base, base + 16);
    }
#pragma unroll
    for (int ns = 0; ns < 4; ++ns) {
      // B 32x16 layout: lane = N col, lane[4] picks K 0-7/16-23 vs 8-15/24-31
      const bf16* base = &Wb[(n0 + ns * 16 + nlo) * 32 + hi * 8];
      bfrag[ns] = frag16(base, base + 16);
    }
#pragma unroll
    for (int ms = 0; ms < 2; ++ms)
#pragma unroll
      for (int ns = 0; ns < 4; ++ns)
        acc[ms][ns] = __builtin_amdgcn_wmma_f32_16x16x32_bf16(
            false, afrag[ms], false, bfrag[ns], (short)0, acc[ms][ns],
            false, false);
  }

  // C/D layout: VGPR v -> row m0+ms*16+hi*8+v, lane -> col n0+ns*16+nlo
#pragma unroll
  for (int ms = 0; ms < 2; ++ms)
#pragma unroll
    for (int ns = 0; ns < 4; ++ns) {
      long col = blockN + n0 + ns * 16 + nlo;
#pragma unroll
      for (int v = 0; v < 8; ++v) {
        long row = blockM + m0 + ms * 16 + hi * 8 + v;
        O[row * (long)N + col] = (OutT)acc[ms][ns][v];
      }
    }
}

// ---------------------------------------------------------------------------
// Fused flash attention.  qkv bf16 [B,T,3C]; y bf16 [B,T,C].
// grid = (T/128, H, B); block = 256 (8 waves); wave owns 16 query rows.
// ---------------------------------------------------------------------------
__global__ __launch_bounds__(256) void attn_kernel(
    const bf16* __restrict__ qkv, bf16* __restrict__ y) {
  const int T = 2048, C = 1024, H = 16, D = 64, F = 3 * C;
  const int qblk = blockIdx.x, h = blockIdx.y, b = blockIdx.z;

  const bf16* qptr = qkv + (long)b * T * F + h * D;
  const bf16* kptr = qptr + C;
  const bf16* vptr = qptr + 2 * C;

  __shared__ bf16 Kt[32 * 64];        // [key][d]
  __shared__ bf16 Vt[64 * 32];        // [d][key] (transposed)
  __shared__ bf16 Pbuf[8 * 16 * 32];  // per-wave P staging

  const int tid = threadIdx.x, wave = tid >> 5, lane = tid & 31;
  const int nlo = lane & 15, hi = lane >> 4;
  const int qrow = qblk * 128 + wave * 16;
  bf16* Pw = &Pbuf[wave * 16 * 32];

  v16bf qfrag[2];
#pragma unroll
  for (int c = 0; c < 2; ++c) {
    const bf16* p = qptr + (long)(qrow + nlo) * F + c * 32 + hi * 8;
    qfrag[c] = frag16(p, p + 16);
  }

  const int key = tid >> 3, koff = (tid & 7) * 8;
  const bf16* gK = &kptr[(long)key * F + koff];
  const bf16* gV = &vptr[(long)key * F + koff];
  uint4 rk = *(const uint4*)gK;   // kb = 0
  uint4 rv = *(const uint4*)gV;

  v8f accO[4] = {};
  float mrow[8], lrow[8];
#pragma unroll
  for (int v = 0; v < 8; ++v) { mrow[v] = -1e30f; lrow[v] = 0.0f; }

  for (int kb = 0; kb < T; kb += 32) {
    __syncthreads();
    *(uint4*)&Kt[key * 64 + koff] = rk;
    {
      union { uint4 u; bf16 h8[8]; } vv;
      vv.u = rv;
#pragma unroll
      for (int i = 0; i < 8; ++i) Vt[(koff + i) * 32 + key] = vv.h8[i];
    }
    __syncthreads();

    if (kb + 32 < T) {
      rk = *(const uint4*)(gK + (long)(kb + 32) * F);
      rv = *(const uint4*)(gV + (long)(kb + 32) * F);
    }
    if (kb + 64 < T) {
      __builtin_prefetch(gK + (long)(kb + 64) * F, 0, 0);
      __builtin_prefetch(gV + (long)(kb + 64) * F, 0, 0);
    }

    // S = Q * K^T
    v8f s0 = {}, s1 = {};
#pragma unroll
    for (int c = 0; c < 2; ++c) {
      const bf16* kb0 = &Kt[(nlo) * 64 + c * 32 + hi * 8];
      s0 = __builtin_amdgcn_wmma_f32_16x16x32_bf16(
          false, qfrag[c], false, frag16(kb0, kb0 + 16), (short)0, s0, false, false);
      const bf16* kb1 = &Kt[(16 + nlo) * 64 + c * 32 + hi * 8];
      s1 = __builtin_amdgcn_wmma_f32_16x16x32_bf16(
          false, qfrag[c], false, frag16(kb1, kb1 + 16), (short)0, s1, false, false);
    }

    // online softmax
    float a0[8], a1[8], rmax[8];
#pragma unroll
    for (int v = 0; v < 8; ++v) {
      a0[v] = s0[v] * 0.125f;  // 1/sqrt(64)
      a1[v] = s1[v] * 0.125f;
      rmax[v] = fmaxf(a0[v], a1[v]);
    }
#pragma unroll
    for (int off = 8; off >= 1; off >>= 1)
#pragma unroll
      for (int v = 0; v < 8; ++v)
        rmax[v] = fmaxf(rmax[v], __shfl_xor(rmax[v], off, 32));

    float p0[8], p1[8], sumr[8], corr[8];
#pragma unroll
    for (int v = 0; v < 8; ++v) {
      float mnew = fmaxf(mrow[v], rmax[v]);
      corr[v] = __expf(mrow[v] - mnew);
      mrow[v] = mnew;
      p0[v] = __expf(a0[v] - mnew);
      p1[v] = __expf(a1[v] - mnew);
      sumr[v] = p0[v] + p1[v];
    }
#pragma unroll
    for (int off = 8; off >= 1; off >>= 1)
#pragma unroll
      for (int v = 0; v < 8; ++v)
        sumr[v] += __shfl_xor(sumr[v], off, 32);
#pragma unroll
    for (int v = 0; v < 8; ++v) lrow[v] = lrow[v] * corr[v] + sumr[v];
#pragma unroll
    for (int dt = 0; dt < 4; ++dt)
#pragma unroll
      for (int v = 0; v < 8; ++v) accO[dt][v] *= corr[v];

    // restage P through per-wave LDS (DS ops in-order per wave)
#pragma unroll
    for (int v = 0; v < 8; ++v) {
      int r = hi * 8 + v;
      Pw[r * 32 + nlo]      = (bf16)p0[v];
      Pw[r * 32 + 16 + nlo] = (bf16)p1[v];
    }
    const bf16* pb = &Pw[nlo * 32 + hi * 8];
    v16bf pfrag = frag16(pb, pb + 16);

    // O += P * V
#pragma unroll
    for (int dt = 0; dt < 4; ++dt) {
      const bf16* vb = &Vt[(dt * 16 + nlo) * 32 + hi * 8];
      accO[dt] = __builtin_amdgcn_wmma_f32_16x16x32_bf16(
          false, pfrag, false, frag16(vb, vb + 16), (short)0, accO[dt],
          false, false);
    }
  }

  float linv[8];
#pragma unroll
  for (int v = 0; v < 8; ++v) linv[v] = 1.0f / lrow[v];
#pragma unroll
  for (int dt = 0; dt < 4; ++dt)
#pragma unroll
    for (int v = 0; v < 8; ++v) {
      long row = (long)b * T + qrow + hi * 8 + v;
      y[row * C + h * D + dt * 16 + nlo] = (bf16)(accO[dt][v] * linv[v]);
    }
}

// ---------------------------------------------------------------------------
extern "C" void kernel_launch(void* const* d_in, const int* in_sizes, int n_in,
                              void* d_out, int out_size, void* d_ws, size_t ws_size,
                              hipStream_t stream) {
  const int B = 2, T = 2048, C = 1024;
  const int M = B * T;         // 4096
  const int F = 3 * C;         // 3072

  const float* x    = (const float*)d_in[0];
  const float* Wqkv = (const float*)d_in[3];
  const float* Wout = (const float*)d_in[4];
  float* out = (float*)d_out;

  bf16* xb   = (bf16*)d_ws;
  bf16* wqb  = xb   + (size_t)M * C;
  bf16* wob  = wqb  + (size_t)F * C;
  bf16* qkvb = wob  + (size_t)C * C;
  bf16* yb   = qkvb + (size_t)M * F;

  f32_to_bf16_kernel<<<(M * C) / 1024, 256, 0, stream>>>(x, xb, M * C);
  f32_to_bf16_kernel<<<(F * C) / 1024, 256, 0, stream>>>(Wqkv, wqb, F * C);
  f32_to_bf16_kernel<<<(C * C) / 1024, 256, 0, stream>>>(Wout, wob, C * C);

  gemm_bf16_kernel<bf16><<<dim3(F / 128, M / 128), 256, 0, stream>>>(
      xb, wqb, qkvb, M, F, C);

  attn_kernel<<<dim3(T / 128, 16, B), 256, 0, stream>>>(qkvb, yb);

  gemm_bf16_kernel<float><<<dim3(C / 128, M / 128), 256, 0, stream>>>(
      yb, wob, out, M, C, C);
}
